// PiGNNLayer_25185688224413
// MI455X (gfx1250) — matrix operand order repaired and checked by reading, not compiled, not run on hardware
//
#include <hip/hip_runtime.h>
#include <hip/hip_bf16.h>

#define ND 50000      // nodes
#define NE 400000     // edges
#define NG 64         // graphs
#define DIM 128       // d_emb
#define EPB 64        // edges per block (4 WMMA M-tiles)

typedef unsigned short u16;
typedef unsigned int   u32;
typedef __attribute__((ext_vector_type(16))) __bf16 v16bf;
typedef __attribute__((ext_vector_type(8)))  float  v8f;

union BF16Frag { v16bf v; u32 u[8]; };

__device__ __forceinline__ u16 f2bf(float f) {
  u32 u = __float_as_uint(f);
  u32 r = (u + 0x7FFFu + ((u >> 16) & 1u)) >> 16;   // round-to-nearest-even
  return (u16)r;
}

__device__ __forceinline__ float gelu_f(float x) {
  return 0.5f * x * (1.0f + erff(x * 0.7071067811865475f));
}

__device__ __forceinline__ v8f wmma_bf16(v16bf a, v16bf b, v8f c) {
  return __builtin_amdgcn_wmma_f32_16x16x32_bf16(false, a, false, b, (short)0, c, false, false);
}

// A fragment (16x32 bf16, M x K) from LDS row-major bf16, row stride `stride`.
// Lanes 0-15: M=lane, K {k0+0..7, k0+16..23}; lanes 16-31: M=lane-16, K {k0+8..15, k0+24..31}.
__device__ __forceinline__ v16bf load_a_lds(const u16* base, int stride, int k0, int lane) {
  int m  = lane & 15;
  int hh = (lane >> 4) << 3;               // 0 or 8
  const u16* r = base + m * stride + k0 + hh;
  uint4 lo = *(const uint4*)(r);           // K = k0+hh   .. +7
  uint4 hi = *(const uint4*)(r + 16);      // K = k0+16+hh.. +7
  BF16Frag f;
  f.u[0]=lo.x; f.u[1]=lo.y; f.u[2]=lo.z; f.u[3]=lo.w;
  f.u[4]=hi.x; f.u[5]=hi.y; f.u[6]=hi.z; f.u[7]=hi.w;
  return f.v;
}

// B fragment (32x16 bf16, K x N) from pre-transposed bf16 weight Wt[dout][din].
// N = ncol + (lane&15); lanes 0-15 hold K {k0+0..7, k0+16..23}, lanes 16-31 hold K {k0+8..15, k0+24..31}.
__device__ __forceinline__ v16bf load_b_glb(const u16* Wt, int din, int ncol, int k0, int lane) {
  int n  = ncol + (lane & 15);
  int ks = k0 + ((lane >> 4) << 3);
  const u16* r = Wt + (size_t)n * din + ks;
  uint4 lo = *(const uint4*)(r);
  uint4 hi = *(const uint4*)(r + 16);
  BF16Frag f;
  f.u[0]=lo.x; f.u[1]=lo.y; f.u[2]=lo.z; f.u[3]=lo.w;
  f.u[4]=hi.x; f.u[5]=hi.y; f.u[6]=hi.z; f.u[7]=hi.w;
  return f.v;
}

// ---------------- weight transpose + fp32 -> bf16 ----------------
__global__ void pignn_wt_convert(const float* __restrict__ W, u16* __restrict__ Wt,
                                 int din, int dout) {
  int i = blockIdx.x * 256 + threadIdx.x;
  if (i >= din * dout) return;
  int k = i / dout, n = i % dout;
  Wt[(size_t)n * din + k] = f2bf(W[i]);
}

// ---------------- fused edge kernel (64 edges / block, B-fragment reuse x4) ----------------
__global__ __launch_bounds__(256)
void pignn_edge(const float* __restrict__ h, const float* __restrict__ e,
                const int* __restrict__ eidx,
                const u16* __restrict__ qWt, const float* __restrict__ qb,
                const u16* __restrict__ kWt, const float* __restrict__ kb,
                const u16* __restrict__ vWt, const float* __restrict__ vb,
                const u16* __restrict__ oWt, const float* __restrict__ ob,
                const u16* __restrict__ e1Wt, const float* __restrict__ e1b,
                const float* __restrict__ eln_g, const float* __restrict__ eln_b,
                const u16* __restrict__ e2Wt, const float* __restrict__ e2b,
                const float* __restrict__ ln1_g, const float* __restrict__ ln1_b,
                float* __restrict__ msg_accum, float* __restrict__ deg,
                float* __restrict__ e_out) {
  __shared__ __align__(16) u16   AeB[EPB][384];   // [hs | e | hd] bf16       48 KB
  __shared__ __align__(16) float e32[EPB][128];   // async-staged fp32 e rows 32 KB
  __shared__ __align__(16) u16   msgB[EPB][128];  //                          16 KB
  __shared__ __align__(16) u16   emB[EPB][128];   //                          16 KB
  __shared__ __align__(16) float t32[EPB][128];   //                          32 KB
  __shared__ float sc[EPB][8];
  __shared__ float at[EPB][8];
  __shared__ int sIdx[EPB], dIdx[EPB];

  const int tid  = threadIdx.x;
  const int lane = tid & 31;
  const int w    = tid >> 5;                    // wave = head = col tile
  const int e0   = blockIdx.x * EPB;

  // ---- async-copy the contiguous fp32 e rows straight into LDS (ASYNCcnt path) ----
  {
    unsigned ldsbase = (unsigned)(size_t)(void*)&e32[0][0];
    const char* gbase = (const char*)(e + (size_t)e0 * DIM);
    #pragma unroll
    for (int it = 0; it < (EPB * DIM * 4) / (256 * 16); ++it) {   // 8 issues x 16B/lane
      unsigned loff = ldsbase + it * 4096 + tid * 16;
      unsigned long long ga = (unsigned long long)(gbase + it * 4096 + tid * 16);
      asm volatile("global_load_async_to_lds_b128 %0, %1, off"
                   :: "v"(loff), "v"(ga) : "memory");
    }
  }

  if (tid < EPB) {
    sIdx[tid] = eidx[e0 + tid];
    dIdx[tid] = eidx[NE + e0 + tid];
  }
  __syncthreads();
  if (tid < EPB) atomicAdd(&deg[dIdx[tid]], 1.0f);

  // gather hs/hd while the e copy is in flight
  for (int i = tid; i < EPB * DIM; i += 256) {
    int m = i >> 7, c = i & 127;
    AeB[m][c]         = f2bf(h[(size_t)sIdx[m] * DIM + c]);          // hs
    AeB[m][2*DIM + c] = f2bf(h[(size_t)dIdx[m] * DIM + c]);          // hd
  }
  asm volatile("s_wait_asynccnt 0x0" ::: "memory");
  __syncthreads();
  for (int i = tid; i < EPB * DIM; i += 256) {
    int m = i >> 7, c = i & 127;
    AeB[m][DIM + c] = f2bf(e32[m][c]);
  }
  __syncthreads();

  // ---- q/k/v projections (head w): load B once per K-step, reuse over 4 M-tiles ----
  v8f qa[4] = {}, ka[4] = {}, va[4] = {};
  #pragma unroll
  for (int kk = 0; kk < 4; ++kk) {
    v16bf bq = load_b_glb(qWt, DIM, w*16, kk*32, lane);
    v16bf bk = load_b_glb(kWt, DIM, w*16, kk*32, lane);
    v16bf bv = load_b_glb(vWt, DIM, w*16, kk*32, lane);
    #pragma unroll
    for (int t = 0; t < 4; ++t) {
      v16bf ahs = load_a_lds(&AeB[t*16][0], 384, kk*32, lane);
      v16bf ahd = load_a_lds(&AeB[t*16][0], 384, 2*DIM + kk*32, lane);
      qa[t] = wmma_bf16(ahd, bq, qa[t]);
      ka[t] = wmma_bf16(ahs, bk, ka[t]);
      va[t] = wmma_bf16(ahs, bv, va[t]);
    }
  }
  {
    int n = w*16 + (lane & 15);
    float bqv = qb[n], bkv = kb[n], bvv = vb[n];
    #pragma unroll
    for (int t = 0; t < 4; ++t)
      #pragma unroll
      for (int r = 0; r < 8; ++r) { qa[t][r] += bqv; ka[t][r] += bkv; va[t][r] += bvv; }
  }

  // ---- per-head score: dot over DH=16 (reduce across 16 lanes) ----
  #pragma unroll
  for (int t = 0; t < 4; ++t) {
    #pragma unroll
    for (int r = 0; r < 8; ++r) {
      float p = qa[t][r] * ka[t][r];
      p += __shfl_xor(p, 1, 16);
      p += __shfl_xor(p, 2, 16);
      p += __shfl_xor(p, 4, 16);
      p += __shfl_xor(p, 8, 16);
      if ((lane & 15) == 0) sc[t*16 + r + 8*(lane>>4)][w] = p * 0.25f;  // / sqrt(16)
    }
  }
  __syncthreads();

  // ---- softmax over heads (per edge) ----
  if (tid < EPB) {
    float mx = -3.4e38f;
    #pragma unroll
    for (int hh = 0; hh < 8; ++hh) mx = fmaxf(mx, sc[tid][hh]);
    float s = 0.f, ex[8];
    #pragma unroll
    for (int hh = 0; hh < 8; ++hh) { ex[hh] = __expf(sc[tid][hh] - mx); s += ex[hh]; }
    float inv = 1.0f / s;
    #pragma unroll
    for (int hh = 0; hh < 8; ++hh) at[tid][hh] = ex[hh] * inv;
  }
  __syncthreads();

  // ---- msg = attn * v, staged as bf16 ----
  #pragma unroll
  for (int t = 0; t < 4; ++t)
    #pragma unroll
    for (int r = 0; r < 8; ++r) {
      int m = t*16 + r + 8*(lane>>4);
      msgB[m][w*16 + (lane & 15)] = f2bf(at[m][w] * va[t][r]);
    }
  __syncthreads();

  // ---- msg @ oW + ob -> scatter-add to nodes ----
  v8f oa[4] = {};
  #pragma unroll
  for (int kk = 0; kk < 4; ++kk) {
    v16bf bo = load_b_glb(oWt, DIM, w*16, kk*32, lane);
    #pragma unroll
    for (int t = 0; t < 4; ++t)
      oa[t] = wmma_bf16(load_a_lds(&msgB[t*16][0], 128, kk*32, lane), bo, oa[t]);
  }
  {
    int n = w*16 + (lane & 15);
    float bov = ob[n];
    #pragma unroll
    for (int t = 0; t < 4; ++t)
      #pragma unroll
      for (int r = 0; r < 8; ++r) {
        int m = t*16 + r + 8*(lane>>4);
        atomicAdd(&msg_accum[(size_t)dIdx[m] * DIM + n], oa[t][r] + bov);
      }
  }

  // ---- edge MLP layer 1: [hs|e|hd] (K=384) @ e1W ----
  v8f ea[4] = {};
  #pragma unroll
  for (int kk = 0; kk < 12; ++kk) {
    v16bf b1f = load_b_glb(e1Wt, 384, w*16, kk*32, lane);
    #pragma unroll
    for (int t = 0; t < 4; ++t)
      ea[t] = wmma_bf16(load_a_lds(&AeB[t*16][0], 384, kk*32, lane), b1f, ea[t]);
  }
  {
    int n = w*16 + (lane & 15);
    float b1 = e1b[n];
    #pragma unroll
    for (int t = 0; t < 4; ++t)
      #pragma unroll
      for (int r = 0; r < 8; ++r) t32[t*16 + r + 8*(lane>>4)][n] = ea[t][r] + b1;
  }
  __syncthreads();

  // ---- LN(eln) + GELU, cooperative (8 rows per wave) ----
  #pragma unroll
  for (int rr = 0; rr < 8; ++rr) {
    int row = w*8 + rr;
    float4 x = *(const float4*)&t32[row][lane*4];
    float s  = x.x + x.y + x.z + x.w;
    float s2 = x.x*x.x + x.y*x.y + x.z*x.z + x.w*x.w;
    #pragma unroll
    for (int mk = 1; mk < 32; mk <<= 1) { s += __shfl_xor(s, mk, 32); s2 += __shfl_xor(s2, mk, 32); }
    float mu   = s * (1.f/128.f);
    float rstd = rsqrtf(s2 * (1.f/128.f) - mu*mu + 1e-5f);
    float xs[4] = {x.x, x.y, x.z, x.w};
    #pragma unroll
    for (int j = 0; j < 4; ++j) {
      int c = lane*4 + j;
      float y = (xs[j] - mu) * rstd * eln_g[c] + eln_b[c];
      emB[row][c] = f2bf(gelu_f(y));
    }
  }
  __syncthreads();

  // ---- edge MLP layer 2 + residual (residual from async-staged e32) ----
  v8f e2a[4] = {};
  #pragma unroll
  for (int kk = 0; kk < 4; ++kk) {
    v16bf b2f = load_b_glb(e2Wt, DIM, w*16, kk*32, lane);
    #pragma unroll
    for (int t = 0; t < 4; ++t)
      e2a[t] = wmma_bf16(load_a_lds(&emB[t*16][0], 128, kk*32, lane), b2f, e2a[t]);
  }
  {
    int n = w*16 + (lane & 15);
    float b2 = e2b[n];
    #pragma unroll
    for (int t = 0; t < 4; ++t)
      #pragma unroll
      for (int r = 0; r < 8; ++r) {
        int m = t*16 + r + 8*(lane>>4);
        t32[m][n] = e32[m][n] + e2a[t][r] + b2;
      }
  }
  __syncthreads();

  // ---- LN1 -> e_new ----
  #pragma unroll
  for (int rr = 0; rr < 8; ++rr) {
    int row = w*8 + rr;
    float4 x = *(const float4*)&t32[row][lane*4];
    float s  = x.x + x.y + x.z + x.w;
    float s2 = x.x*x.x + x.y*x.y + x.z*x.z + x.w*x.w;
    #pragma unroll
    for (int mk = 1; mk < 32; mk <<= 1) { s += __shfl_xor(s, mk, 32); s2 += __shfl_xor(s2, mk, 32); }
    float mu   = s * (1.f/128.f);
    float rstd = rsqrtf(s2 * (1.f/128.f) - mu*mu + 1e-5f);
    float xs[4] = {x.x, x.y, x.z, x.w};
    #pragma unroll
    for (int j = 0; j < 4; ++j) {
      int c = lane*4 + j;
      e_out[(size_t)(e0 + row) * DIM + c] = (xs[j] - mu) * rstd * ln1_g[c] + ln1_b[c];
    }
  }
}

// ---------------- node kernel ----------------
__global__ __launch_bounds__(256)
void pignn_node(const float* __restrict__ h,
                const float* __restrict__ msg_accum, const float* __restrict__ deg,
                const int* __restrict__ batch_idx,
                const u16* __restrict__ n1Wt, const float* __restrict__ n1b,
                const float* __restrict__ nln_g, const float* __restrict__ nln_b,
                const u16* __restrict__ n2Wt, const float* __restrict__ n2b,
                const float* __restrict__ ln2_g, const float* __restrict__ ln2_b,
                float* __restrict__ h_out, float* __restrict__ gsum, float* __restrict__ cnt) {
  __shared__ __align__(16) u16  hinB[16][256];   // [msg_n | h] bf16
  __shared__ __align__(16) u16  nmB[16][256];
  __shared__ __align__(16) float t32[16][256];
  __shared__ float degl[16];
  __shared__ int   bx[16];

  const int tid = threadIdx.x, lane = tid & 31, w = tid >> 5;
  const int n0 = blockIdx.x * 16;

  if (tid < 16) {
    degl[tid] = fmaxf(deg[n0 + tid], 1.0f);
    bx[tid]   = batch_idx[n0 + tid];
    atomicAdd(&cnt[bx[tid]], 1.0f);
  }
  __syncthreads();
  for (int i = tid; i < 16 * DIM; i += 256) {
    int m = i >> 7, c = i & 127;
    hinB[m][c]       = f2bf(msg_accum[(size_t)(n0 + m) * DIM + c] / degl[m]);
    hinB[m][DIM + c] = f2bf(h[(size_t)(n0 + m) * DIM + c]);
  }
  __syncthreads();

  // ---- n1: K=256, dout=256 (two col tiles per wave) ----
  #pragma unroll
  for (int tt = 0; tt < 2; ++tt) {
    int t = w + 8*tt;
    v8f acc = {};
    #pragma unroll
    for (int kk = 0; kk < 8; ++kk)
      acc = wmma_bf16(load_a_lds(&hinB[0][0], 256, kk*32, lane),
                      load_b_glb(n1Wt, 256, t*16, kk*32, lane), acc);
    int n = t*16 + (lane & 15);
    float b1 = n1b[n];
    #pragma unroll
    for (int r = 0; r < 8; ++r) t32[r + 8*(lane>>4)][n] = acc[r] + b1;
  }
  __syncthreads();

  // ---- LN(nln, 256) + GELU ----
  #pragma unroll
  for (int rr = 0; rr < 2; ++rr) {
    int row = w*2 + rr;
    float4 xa = *(const float4*)&t32[row][lane*4];
    float4 xb = *(const float4*)&t32[row][128 + lane*4];
    float s  = xa.x+xa.y+xa.z+xa.w + xb.x+xb.y+xb.z+xb.w;
    float s2 = xa.x*xa.x+xa.y*xa.y+xa.z*xa.z+xa.w*xa.w
             + xb.x*xb.x+xb.y*xb.y+xb.z*xb.z+xb.w*xb.w;
    #pragma unroll
    for (int mk = 1; mk < 32; mk <<= 1) { s += __shfl_xor(s, mk, 32); s2 += __shfl_xor(s2, mk, 32); }
    float mu   = s * (1.f/256.f);
    float rstd = rsqrtf(s2 * (1.f/256.f) - mu*mu + 1e-5f);
    float xs[8] = {xa.x, xa.y, xa.z, xa.w, xb.x, xb.y, xb.z, xb.w};
    #pragma unroll
    for (int j = 0; j < 8; ++j) {
      int c = (j < 4) ? (lane*4 + j) : (128 + lane*4 + j - 4);
      float y = (xs[j] - mu) * rstd * nln_g[c] + nln_b[c];
      nmB[row][c] = f2bf(gelu_f(y));
    }
  }
  __syncthreads();

  // ---- n2: K=256, dout=128 + residual ----
  v8f a2 = {};
  #pragma unroll
  for (int kk = 0; kk < 8; ++kk)
    a2 = wmma_bf16(load_a_lds(&nmB[0][0], 256, kk*32, lane),
                   load_b_glb(n2Wt, 256, w*16, kk*32, lane), a2);
  {
    int n = w*16 + (lane & 15);
    float b2 = n2b[n];
    #pragma unroll
    for (int r = 0; r < 8; ++r) {
      int m = r + 8*(lane>>4);
      t32[m][n] = h[(size_t)(n0 + m) * DIM + n] + a2[r] + b2;
    }
  }
  __syncthreads();

  // ---- LN2 -> h_new, plus graph-sum atomics ----
  #pragma unroll
  for (int rr = 0; rr < 2; ++rr) {
    int row = w*2 + rr;
    float4 x = *(const float4*)&t32[row][lane*4];
    float s  = x.x + x.y + x.z + x.w;
    float s2 = x.x*x.x + x.y*x.y + x.z*x.z + x.w*x.w;
    #pragma unroll
    for (int mk = 1; mk < 32; mk <<= 1) { s += __shfl_xor(s, mk, 32); s2 += __shfl_xor(s2, mk, 32); }
    float mu   = s * (1.f/128.f);
    float rstd = rsqrtf(s2 * (1.f/128.f) - mu*mu + 1e-5f);
    float xs[4] = {x.x, x.y, x.z, x.w};
    #pragma unroll
    for (int j = 0; j < 4; ++j) {
      int c = lane*4 + j;
      float hn = (xs[j] - mu) * rstd * ln2_g[c] + ln2_b[c];
      h_out[(size_t)(n0 + row) * DIM + c] = hn;
      atomicAdd(&gsum[(size_t)bx[row] * DIM + c], hn);
    }
  }
}

// ---------------- gate kernel (64 graphs, 4 blocks of 16) ----------------
__global__ __launch_bounds__(256)
void pignn_gate(const float* __restrict__ gsum, const float* __restrict__ cnt,
                const u16* __restrict__ g1Wt, const float* __restrict__ g1b,
                const float* __restrict__ gln_g, const float* __restrict__ gln_b,
                const u16* __restrict__ g2Wt, const float* __restrict__ g2b,
                float* __restrict__ gates) {
  __shared__ __align__(16) u16  hgB[16][128];
  __shared__ __align__(16) u16  gmB[16][128];
  __shared__ __align__(16) float t32[16][128];
  const int tid = threadIdx.x, lane = tid & 31, w = tid >> 5;
  const int g0 = blockIdx.x * 16;

  for (int i = tid; i < 16 * DIM; i += 256) {
    int m = i >> 7, c = i & 127;
    hgB[m][c] = f2bf(gsum[(size_t)(g0 + m) * DIM + c] / fmaxf(cnt[g0 + m], 1.0f));
  }
  __syncthreads();

  v8f a1 = {};
  #pragma unroll
  for (int kk = 0; kk < 4; ++kk)
    a1 = wmma_bf16(load_a_lds(&hgB[0][0], 128, kk*32, lane),
                   load_b_glb(g1Wt, DIM, w*16, kk*32, lane), a1);
  {
    int n = w*16 + (lane & 15);
    float b1 = g1b[n];
    #pragma unroll
    for (int r = 0; r < 8; ++r) t32[r + 8*(lane>>4)][n] = a1[r] + b1;
  }
  __syncthreads();

  #pragma unroll
  for (int rr = 0; rr < 2; ++rr) {
    int row = w*2 + rr;
    float4 x = *(const float4*)&t32[row][lane*4];
    float s  = x.x + x.y + x.z + x.w;
    float s2 = x.x*x.x + x.y*x.y + x.z*x.z + x.w*x.w;
    #pragma unroll
    for (int mk = 1; mk < 32; mk <<= 1) { s += __shfl_xor(s, mk, 32); s2 += __shfl_xor(s2, mk, 32); }
    float mu   = s * (1.f/128.f);
    float rstd = rsqrtf(s2 * (1.f/128.f) - mu*mu + 1e-5f);
    float xs[4] = {x.x, x.y, x.z, x.w};
    #pragma unroll
    for (int j = 0; j < 4; ++j) {
      int c = lane*4 + j;
      float y = (xs[j] - mu) * rstd * gln_g[c] + gln_b[c];
      gmB[row][c] = f2bf(gelu_f(y));
    }
  }
  __syncthreads();

  v8f a2 = {};
  #pragma unroll
  for (int kk = 0; kk < 4; ++kk)
    a2 = wmma_bf16(load_a_lds(&gmB[0][0], 128, kk*32, lane),
                   load_b_glb(g2Wt, DIM, w*16, kk*32, lane), a2);
  {
    int n = w*16 + (lane & 15);
    float b2 = g2b[n];
    #pragma unroll
    for (int r = 0; r < 8; ++r) {
      int m = r + 8*(lane>>4);
      float x = a2[r] + b2;
      gates[(size_t)(g0 + m) * DIM + n] = 1.0f / (1.0f + __expf(-x));
    }
  }
}

// ---------------- apply gates ----------------
__global__ void pignn_apply(float* __restrict__ hout, const float* __restrict__ gates,
                            const int* __restrict__ batch_idx) {
  size_t i = (size_t)blockIdx.x * 256 + threadIdx.x;
  if (i >= (size_t)ND * DIM) return;
  int node = (int)(i >> 7), c = (int)(i & 127);
  hout[i] *= gates[(size_t)batch_idx[node] * DIM + c];
}

// ---------------- host ----------------
extern "C" void kernel_launch(void* const* d_in, const int* in_sizes, int n_in,
                              void* d_out, int out_size, void* d_ws, size_t ws_size,
                              hipStream_t stream) {
  (void)in_sizes; (void)n_in; (void)out_size; (void)ws_size;

  const float* h     = (const float*)d_in[0];
  const float* e     = (const float*)d_in[1];
  const int*   eidx  = (const int*)d_in[2];
  const int*   bidx  = (const int*)d_in[3];
  const float* qW = (const float*)d_in[4],  *qb = (const float*)d_in[5];
  const float* kW = (const float*)d_in[6],  *kb = (const float*)d_in[7];
  const float* vW = (const float*)d_in[8],  *vb = (const float*)d_in[9];
  const float* oW = (const float*)d_in[10], *ob = (const float*)d_in[11];
  const float* e1W = (const float*)d_in[12], *e1b = (const float*)d_in[13];
  const float* eln_g = (const float*)d_in[14], *eln_b = (const float*)d_in[15];
  const float* e2W = (const float*)d_in[16], *e2b = (const float*)d_in[17];
  const float* n1W = (const float*)d_in[18], *n1b = (const float*)d_in[19];
  const float* nln_g = (const float*)d_in[20], *nln_b = (const float*)d_in[21];
  const float* n2W = (const float*)d_in[22], *n2b = (const float*)d_in[23];
  const float* g1W = (const float*)d_in[24], *g1b = (const float*)d_in[25];
  const float* gln_g = (const float*)d_in[26], *gln_b = (const float*)d_in[27];
  const float* g2W = (const float*)d_in[28], *g2b = (const float*)d_in[29];
  const float* ln1_g = (const float*)d_in[30], *ln1_b = (const float*)d_in[31];
  const float* ln2_g = (const float*)d_in[32], *ln2_b = (const float*)d_in[33];

  // workspace carve-up (256B aligned)
  size_t off = 0;
  auto take = [&](size_t bytes) -> void* {
    void* p = (char*)d_ws + off;
    off += (bytes + 255) & ~(size_t)255;
    return p;
  };
  float* msg_accum = (float*)take((size_t)ND * DIM * 4);
  float* deg       = (float*)take((size_t)ND * 4);
  float* gsum      = (float*)take((size_t)NG * DIM * 4);
  float* cnt       = (float*)take((size_t)NG * 4);
  float* gates     = (float*)take((size_t)NG * DIM * 4);
  u16* qWt  = (u16*)take((size_t)128*128*2);
  u16* kWt  = (u16*)take((size_t)128*128*2);
  u16* vWt  = (u16*)take((size_t)128*128*2);
  u16* oWt  = (u16*)take((size_t)128*128*2);
  u16* e1Wt = (u16*)take((size_t)384*128*2);
  u16* e2Wt = (u16*)take((size_t)128*128*2);
  u16* n1Wt = (u16*)take((size_t)256*256*2);
  u16* n2Wt = (u16*)take((size_t)256*128*2);
  u16* g1Wt = (u16*)take((size_t)128*128*2);
  u16* g2Wt = (u16*)take((size_t)128*128*2);

  hipMemsetAsync(msg_accum, 0, (size_t)ND * DIM * 4, stream);
  hipMemsetAsync(deg,       0, (size_t)ND * 4,       stream);
  hipMemsetAsync(gsum,      0, (size_t)NG * DIM * 4, stream);
  hipMemsetAsync(cnt,       0, (size_t)NG * 4,       stream);

  auto conv = [&](const float* W, u16* Wt, int din, int dout) {
    int n = din * dout;
    pignn_wt_convert<<<(n + 255) / 256, 256, 0, stream>>>(W, Wt, din, dout);
  };
  conv(qW, qWt, 128, 128);  conv(kW, kWt, 128, 128);
  conv(vW, vWt, 128, 128);  conv(oW, oWt, 128, 128);
  conv(e1W, e1Wt, 384, 128); conv(e2W, e2Wt, 128, 128);
  conv(n1W, n1Wt, 256, 256); conv(n2W, n2Wt, 256, 128);
  conv(g1W, g1Wt, 128, 128); conv(g2W, g2Wt, 128, 128);

  float* out   = (float*)d_out;
  float* e_out = out + (size_t)ND * DIM;

  pignn_edge<<<NE / EPB, 256, 0, stream>>>(
      h, e, eidx,
      qWt, qb, kWt, kb, vWt, vb, oWt, ob,
      e1Wt, e1b, eln_g, eln_b, e2Wt, e2b, ln1_g, ln1_b,
      msg_accum, deg, e_out);

  pignn_node<<<ND / 16, 256, 0, stream>>>(
      h, msg_accum, deg, bidx,
      n1Wt, n1b, nln_g, nln_b, n2Wt, n2b, ln2_g, ln2_b,
      out, gsum, cnt);

  pignn_gate<<<NG / 16, 256, 0, stream>>>(
      gsum, cnt, g1Wt, g1b, gln_g, gln_b, g2Wt, g2b, gates);

  pignn_apply<<<(ND * DIM + 255) / 256, 256, 0, stream>>>(out, gates, bidx);
}
